// MambaBlock_28716151341388
// MI455X (gfx1250) — compile-verified, matrix operand order checked by reading, not run on hardware
//
#include <hip/hip_runtime.h>
#include <math.h>

// ---------------------------------------------------------------------------
// Types for CDNA5 WMMA (wave32, 16x16x32 bf16 -> f32)
// ---------------------------------------------------------------------------
typedef __bf16 bf16_t;
typedef __attribute__((ext_vector_type(16))) __bf16 v16bf;
typedef __attribute__((ext_vector_type(8)))  __bf16 v8bf;
typedef __attribute__((ext_vector_type(8)))  float  v8f;

#define D_MODEL 768
#define D_STATE 16
#define D_CONV  4
#define D_INNER 1536

// ---------------------------------------------------------------------------
// CDNA5 async global->LDS copy (ASYNCcnt path), via inline asm so it is
// identical on both toolchains. Each active lane copies 16 bytes.
// ---------------------------------------------------------------------------
__device__ __forceinline__ void async_copy_b128(const void* gsrc, unsigned lds_off) {
    asm volatile("global_load_async_to_lds_b128 %0, %1, off"
                 :: "v"(lds_off), "v"(gsrc)
                 : "memory");
}
__device__ __forceinline__ void wait_async0() {
    asm volatile("s_wait_asynccnt 0x0" ::: "memory");
}

// ---------------------------------------------------------------------------
// Helpers
// ---------------------------------------------------------------------------
__device__ __forceinline__ float block_reduce_sum_256(float v, float* red) {
    const int tid = threadIdx.x;
    red[tid] = v;
    __syncthreads();
    for (int s = 128; s > 0; s >>= 1) {
        if (tid < s) red[tid] += red[tid + s];
        __syncthreads();
    }
    float r = red[0];
    __syncthreads();
    return r;
}

__device__ __forceinline__ float silu_f(float v) {
    return v / (1.0f + __expf(-v));
}

// ---------------------------------------------------------------------------
// Scale pipeline for bitlinear: mean(|W|) -> scale
// ws float slots: [0]=acc_inp [1]=acc_out [2]=scale_inp [3]=scale_out
// ---------------------------------------------------------------------------
__global__ void init_scales_kernel(float* sc) {
    if (threadIdx.x == 0) { sc[0] = 0.0f; sc[1] = 0.0f; }
}

__global__ void __launch_bounds__(256) absmean_partial_kernel(
        const float* __restrict__ W, int n, float* __restrict__ acc) {
    __shared__ float red[256];
    float s = 0.0f;
    for (int i = blockIdx.x * blockDim.x + threadIdx.x; i < n;
         i += gridDim.x * blockDim.x)
        s += fabsf(W[i]);
    s = block_reduce_sum_256(s, red);
    if (threadIdx.x == 0) atomicAdd(acc, s);
}

__global__ void finalize_scales_kernel(float* sc, int n1, int n2) {
    if (threadIdx.x == 0) {
        sc[2] = fmaxf(sc[0] / (float)n1, 1e-5f);
        sc[3] = fmaxf(sc[1] / (float)n2, 1e-5f);
    }
}

// Ternary-quantize weights to bf16:  round(clip(W/scale, -1, 1))
__global__ void __launch_bounds__(256) quantize_weights_kernel(
        const float* __restrict__ W, const float* __restrict__ sc, int which,
        bf16_t* __restrict__ out, int n) {
    int i = blockIdx.x * blockDim.x + threadIdx.x;
    if (i >= n) return;
    float s = sc[2 + which];
    float w = W[i] / s;
    w = fminf(1.0f, fmaxf(-1.0f, w));
    out[i] = (bf16_t)rintf(w);
}

// ---------------------------------------------------------------------------
// (optionally chained) RMSNorm over rows -> bf16 activations for WMMA
// two_stage==1: xn = rmsnorm(rmsnorm(x,w1), w2) ; else xn = rmsnorm(x,w2)
// ---------------------------------------------------------------------------
__global__ void __launch_bounds__(256) rmsnorm_quant_rows_kernel(
        const float* __restrict__ X, const float* __restrict__ w1,
        const float* __restrict__ w2, bf16_t* __restrict__ out,
        int K, int two_stage) {
    __shared__ float sx[D_INNER];   // max K = 1536
    __shared__ float red[256];
    const int row = blockIdx.x;
    const int tid = threadIdx.x;
    const float* xr = X + (size_t)row * K;

    float s = 0.0f;
    for (int k = tid; k < K; k += 256) {
        float v = xr[k];
        sx[k] = v;
        s += v * v;
    }
    s = block_reduce_sum_256(s, red);
    float rs = rsqrtf(s / (float)K + 1e-6f);

    if (two_stage) {
        float s2 = 0.0f;
        for (int k = tid; k < K; k += 256) {
            float h = sx[k] * rs * w1[k];
            sx[k] = h;
            s2 += h * h;
        }
        s2 = block_reduce_sum_256(s2, red);
        rs = rsqrtf(s2 / (float)K + 1e-6f);
    }
    bf16_t* orow = out + (size_t)row * K;
    for (int k = tid; k < K; k += 256)
        orow[k] = (bf16_t)(sx[k] * rs * w2[k]);
}

// ---------------------------------------------------------------------------
// WMMA GEMM:  C[M,N] = A[M,K](bf16) @ Wq[N,K]^T(bf16) * scale (+ residual)
//
// Block = 256 threads (8 waves) -> 128x64 C tile. The 64-col B tile is shared
// by all 8 waves: per K-step it is staged into LDS with the CDNA5 async
// global->LDS path (global_load_async_to_lds_b128 / s_wait_asynccnt), double
// buffered so the next tile's copy overlaps the current tile's WMMAs.
// A fragments are loaded directly (each wave owns distinct rows; L2-hot),
// with global_prefetch_b8 on the next K tile.
//
// Fragment layout per CDNA5 ISA (16-bit A/B 16x32): lane&15 = row,
// lane>=16 selects K halves {8..15,24..31} vs {0..7,16..23}.
// LDS B rows use an 80B pitch: 16B-aligned and bank-conflict-free for the
// 16-lane ds_load_b128 fragment reads.
// ---------------------------------------------------------------------------
#define BPITCH 80
#define BBUF   (64 * BPITCH)

__global__ void __launch_bounds__(256) gemm_bf16_wmma_kernel(
        const bf16_t* __restrict__ A, const bf16_t* __restrict__ Bq,
        const float* __restrict__ scale_ptr, const float* __restrict__ residual,
        float* __restrict__ C, int M, int N, int K) {
    __shared__ __align__(16) unsigned char sB[2][BBUF];

    const int tid  = threadIdx.x;
    const int lane = tid & 31;
    const int wv   = tid >> 5;                 // wave 0..7 -> 16-row slice
    const int nBlk = N >> 6;
    const int bm   = blockIdx.x / nBlk;        // 128-row block tile
    const int bn   = blockIdx.x % nBlk;        // 64-col block tile
    const int n0   = bn << 6;
    const int m0   = (bm << 7) + (wv << 4);

    // cooperative B-tile copy: 256 threads = 64 rows x 4 x 16B chunks
    const int crow = tid >> 2;
    const int cchk = tid & 3;
    const bf16_t* gB = Bq + (size_t)(n0 + crow) * K + cchk * 8;
    const unsigned lB =
        (unsigned)(uintptr_t)(&sB[0][0] + crow * BPITCH + cchk * 16);

    // WMMA fragment addressing
    const int row   = lane & 15;
    const int khalf = (lane >> 4) & 1;
    const bf16_t* arow = A + (size_t)(m0 + row) * K + khalf * 8;
    const unsigned char* fB = &sB[0][0] + row * BPITCH + khalf * 16;

    v8f acc[4] = {v8f{}, v8f{}, v8f{}, v8f{}};

    async_copy_b128(gB, lB);                   // prime buffer 0
    int cur = 0;

    for (int k0 = 0; k0 < K; k0 += 32) {
        wait_async0();                         // my 16B chunk has landed
        __syncthreads();                       // whole tile has landed
        if (k0 + 32 < K) {                     // overlap next copy w/ compute
            async_copy_b128(gB + (k0 + 32), lB + (cur ^ 1) * BBUF);
            __builtin_prefetch(arow + k0 + 32, 0, 1);
        }

        v8bf alo = *(const v8bf*)(arow + k0);
        v8bf ahi = *(const v8bf*)(arow + k0 + 16);
        v16bf af = __builtin_shufflevector(alo, ahi,
                     0,1,2,3,4,5,6,7,8,9,10,11,12,13,14,15);

        const unsigned char* bbase = fB + cur * BBUF;
#pragma unroll
        for (int j = 0; j < 4; ++j) {
            const unsigned char* bp = bbase + (16 * j) * BPITCH;
            v8bf blo = *(const v8bf*)(bp);
            v8bf bhi = *(const v8bf*)(bp + 32);
            v16bf bfr = __builtin_shufflevector(blo, bhi,
                          0,1,2,3,4,5,6,7,8,9,10,11,12,13,14,15);
            acc[j] = __builtin_amdgcn_wmma_f32_16x16x32_bf16(
                         false, af, false, bfr, (short)0, acc[j], false, false);
        }
        cur ^= 1;
        // no trailing barrier needed: the WMMAs above force s_wait_dscnt on
        // this buffer's reads before the next iteration's top barrier.
    }

    const float s  = scale_ptr ? scale_ptr[0] : 1.0f;
    const int   nc = lane & 15;
    const int   mb = m0 + ((lane >> 4) << 3);  // +8 rows for upper half-wave
#pragma unroll
    for (int j = 0; j < 4; ++j) {
        const int col = n0 + 16 * j + nc;
#pragma unroll
        for (int r = 0; r < 8; ++r) {
            size_t idx = (size_t)(mb + r) * N + col;
            float v = acc[j][r] * s;
            if (residual) v += residual[idx];
            C[idx] = v;
        }
    }
}

// ---------------------------------------------------------------------------
// Causal depthwise conv (taps=4) + bias + SiLU.  Input = x-half of xz.
// ---------------------------------------------------------------------------
__global__ void __launch_bounds__(256) dwconv_silu_kernel(
        const float* __restrict__ xz, const float* __restrict__ cw,
        const float* __restrict__ cb, float* __restrict__ out,
        int Bb, int L) {
    int idx = blockIdx.x * blockDim.x + threadIdx.x;
    int total = Bb * L * D_INNER;
    if (idx >= total) return;
    int d = idx % D_INNER;
    int l = (idx / D_INNER) % L;
    int b = idx / (D_INNER * L);

    float acc = cb[d];
#pragma unroll
    for (int j = 0; j < D_CONV; ++j) {
        int ls = l - (D_CONV - 1) + j;
        if (ls >= 0)
            acc += xz[((size_t)b * L + ls) * (2 * D_INNER) + d] * cw[d * D_CONV + j];
    }
    out[idx] = silu_f(acc);
}

// ---------------------------------------------------------------------------
// x-proj GEMV: dbc[row, 0..32] = xconv_row @ xproj_W^T   (33 outputs/row)
// 288 threads = 9 waves; group of 8 lanes per output, shfl-reduced.
// ---------------------------------------------------------------------------
__global__ void __launch_bounds__(288) xproj_kernel(
        const float* __restrict__ xc, const float* __restrict__ W,
        float* __restrict__ dbc) {
    __shared__ float sx[D_INNER];
    const int r = blockIdx.x;
    const int tid = threadIdx.x;
    const float* xr = xc + (size_t)r * D_INNER;
    for (int k = tid; k < D_INNER; k += 288) sx[k] = xr[k];
    __syncthreads();

    const int g  = tid >> 3;
    const int l8 = tid & 7;
    if (g < 2 * D_STATE + 1) {
        const float* wr = W + (size_t)g * D_INNER;
        float s = 0.0f;
        for (int k = l8; k < D_INNER; k += 8) s += sx[k] * wr[k];
        s += __shfl_down(s, 4, 8);
        s += __shfl_down(s, 2, 8);
        s += __shfl_down(s, 1, 8);
        if (l8 == 0) dbc[(size_t)r * (2 * D_STATE + 1) + g] = s;
    }
}

// ---------------------------------------------------------------------------
// Selective scan. 16 lanes own the 16 states of one (b,d) channel; the
// per-step y reduction is 4 shfl_xor ops in a 16-lane segment (wave32).
// Fuses dt projection + softplus, +x*D skip, and *silu(z) gate.
// yout may alias xconv (same index read->write inside one wave step).
// ---------------------------------------------------------------------------
__global__ void __launch_bounds__(256) scan_kernel(
        const float* __restrict__ dbc, const float* __restrict__ dtW,
        const float* __restrict__ dtb, const float* __restrict__ A_log,
        const float* __restrict__ Dp, const float* xconv,
        const float* __restrict__ xz, float* yout, int Bb, int L) {
    int gid = blockIdx.x * blockDim.x + threadIdx.x;
    int n   = gid & 15;
    int ch  = gid >> 4;
    if (ch >= Bb * D_INNER) return;
    int b = ch / D_INNER;
    int d = ch % D_INNER;

    const float An   = -__expf(A_log[d * D_STATE + n]);
    const float dtw  = dtW[d];
    const float dtbv = dtb[d];
    const float Dd   = Dp[d];
    float h = 0.0f;

    const float* dbc_b = dbc + (size_t)b * L * (2 * D_STATE + 1);
    const float* xc_b  = xconv + ((size_t)b * L) * D_INNER + d;
    const float* z_b   = xz + ((size_t)b * L) * (2 * D_INNER) + D_INNER + d;
    float*       y_b   = yout + ((size_t)b * L) * D_INNER + d;

    for (int t = 0; t < L; ++t) {
        const float* rowp = dbc_b + (size_t)t * (2 * D_STATE + 1);
        float draw = rowp[0];
        float Bt   = rowp[1 + n];
        float Ct   = rowp[1 + D_STATE + n];
        float xt   = xc_b[(size_t)t * D_INNER];
        float zt   = z_b[(size_t)t * (2 * D_INNER)];

        float u  = draw * dtw + dtbv;
        float sp = (u > 20.0f) ? u : log1pf(__expf(u));   // softplus

        h = __expf(sp * An) * h + (sp * xt) * Bt;
        float yn = h * Ct;
        yn += __shfl_xor(yn, 1, 16);
        yn += __shfl_xor(yn, 2, 16);
        yn += __shfl_xor(yn, 4, 16);
        yn += __shfl_xor(yn, 8, 16);

        if (n == 0) {
            float yv = yn + xt * Dd;
            y_b[(size_t)t * D_INNER] = yv * silu_f(zt);
        }
    }
}

// ---------------------------------------------------------------------------
// Launch
// ---------------------------------------------------------------------------
extern "C" void kernel_launch(void* const* d_in, const int* in_sizes, int n_in,
                              void* d_out, int out_size, void* d_ws, size_t ws_size,
                              hipStream_t stream) {
    const float* x          = (const float*)d_in[0];
    const float* norm_w     = (const float*)d_in[1];
    const float* inp_norm_w = (const float*)d_in[2];
    const float* inp_W      = (const float*)d_in[3];
    const float* conv_w     = (const float*)d_in[4];
    const float* conv_b     = (const float*)d_in[5];
    const float* xproj_W    = (const float*)d_in[6];
    const float* dt_W       = (const float*)d_in[7];
    const float* dt_b       = (const float*)d_in[8];
    const float* A_log      = (const float*)d_in[9];
    const float* Dp         = (const float*)d_in[10];
    const float* out_norm_w = (const float*)d_in[11];
    const float* out_W      = (const float*)d_in[12];
    float* out = (float*)d_out;

    const int BL = in_sizes[0] / D_MODEL;   // B * L (4096)
    const int L  = 2048;
    const int Bb = BL / L;

    // -------- workspace layout (aliased; ~87 MB) --------
    float*  sc    = (float*)d_ws;                           // scale slots
    char*   p     = (char*)d_ws + 256;
    bf16_t* xn1   = (bf16_t*)p;  p += (size_t)BL * D_MODEL * 2;
    bf16_t* Wq1   = (bf16_t*)p;  p += (size_t)2 * D_INNER * D_MODEL * 2;
    float*  xz    = (float*)p;   p += (size_t)BL * 2 * D_INNER * 4;
    float*  xconv = (float*)p;   p += (size_t)BL * D_INNER * 4;
    float*  dbc   = (float*)p;   p += (size_t)BL * (2 * D_STATE + 1) * 4;
    float*  yscan = xconv;                          // scan runs in place
    bf16_t* yn    = (bf16_t*)xz;                    // xz dead after scan
    bf16_t* Wq2   = (bf16_t*)((char*)xz + (size_t)BL * D_INNER * 2);

    const int n_inpW = 2 * D_INNER * D_MODEL;   // 2359296
    const int n_outW = D_MODEL * D_INNER;       // 1179648

    // 1) bitlinear scales
    init_scales_kernel<<<1, 32, 0, stream>>>(sc);
    absmean_partial_kernel<<<64, 256, 0, stream>>>(inp_W, n_inpW, sc + 0);
    absmean_partial_kernel<<<64, 256, 0, stream>>>(out_W, n_outW, sc + 1);
    finalize_scales_kernel<<<1, 32, 0, stream>>>(sc, n_inpW, n_outW);

    // 2) chained rmsnorm -> bf16 A, ternary weights -> bf16 B
    rmsnorm_quant_rows_kernel<<<BL, 256, 0, stream>>>(
        x, norm_w, inp_norm_w, xn1, D_MODEL, 1);
    quantize_weights_kernel<<<(n_inpW + 255) / 256, 256, 0, stream>>>(
        inp_W, sc, 0, Wq1, n_inpW);

    // 3) GEMM1: xz = xn1 @ Wq1^T * scale   (M=BL, N=3072, K=768)
    {
        int blocks = (BL >> 7) * ((2 * D_INNER) >> 6);
        gemm_bf16_wmma_kernel<<<blocks, 256, 0, stream>>>(
            xn1, Wq1, sc + 2, nullptr, xz, BL, 2 * D_INNER, D_MODEL);
    }

    // 4) depthwise causal conv + SiLU on x half
    {
        int total = BL * D_INNER;
        dwconv_silu_kernel<<<(total + 255) / 256, 256, 0, stream>>>(
            xz, conv_w, conv_b, xconv, Bb, L);
    }

    // 5) x-proj: dbc = xconv @ xproj_W^T
    xproj_kernel<<<BL, 288, 0, stream>>>(xconv, xproj_W, dbc);

    // 6) selective scan (fused dt/softplus, skip, silu(z) gate); in place
    {
        int total = Bb * D_INNER * D_STATE;
        scan_kernel<<<(total + 255) / 256, 256, 0, stream>>>(
            dbc, dt_W, dt_b, A_log, Dp, xconv, xz, yscan, Bb, L);
    }

    // 7) out bitlinear: rmsnorm -> bf16, quantize out_W, WMMA GEMM + residual
    rmsnorm_quant_rows_kernel<<<BL, 256, 0, stream>>>(
        yscan, nullptr, out_norm_w, yn, D_INNER, 0);
    quantize_weights_kernel<<<(n_outW + 255) / 256, 256, 0, stream>>>(
        out_W, sc, 1, Wq2, n_outW);
    {
        int blocks = (BL >> 7) * (D_MODEL >> 6);
        gemm_bf16_wmma_kernel<<<blocks, 256, 0, stream>>>(
            yn, Wq2, sc + 3, x, out, BL, D_MODEL, D_INNER);
    }
}
